// contrastive_loss_78288663871843
// MI455X (gfx1250) — matrix operand order repaired and checked by reading, not compiled, not run on hardware
//
#include <hip/hip_runtime.h>
#include <stdint.h>

#define NROW  4096
#define DIM   128
#define TWO_N 8192
#define COL_SPLIT        8
#define COL_TILES        (TWO_N / 16)                 // 512
#define TILES_PER_SPLIT  (COL_TILES / COL_SPLIT)      // 64
#define ROW_BLOCKS       (TWO_N / 256)                // 32 (8 waves x 32 rows)

typedef __attribute__((ext_vector_type(16))) _Float16 v16h;
typedef __attribute__((ext_vector_type(8)))  _Float16 v8h;
typedef __attribute__((ext_vector_type(4)))  _Float16 v4h;
typedef __attribute__((ext_vector_type(8)))  float    v8f;
typedef __attribute__((ext_vector_type(4)))  float    v4f;

// ---- CDNA5 async global->LDS copy (ASYNCcnt path), per ISA 15.18.3 ---------
__device__ __forceinline__ void async_copy_b128(uint32_t lds_byte_off,
                                                const void* gaddr) {
    // LDS address comes from the vdst VGPR; flat shared ptr low 32 bits == LDS offset
    asm volatile("global_load_async_to_lds_b128 %0, %1, off"
                 :: "v"(lds_byte_off), "v"(gaddr) : "memory");
}
__device__ __forceinline__ void wait_async_zero() {
    asm volatile("s_wait_asynccnt 0x0" ::: "memory");
}

// ---------------------------------------------------------------------------
// Kernel 1: L2-normalize rows of [student; teacher].
// Writes reps[8192][128] (row-major, feeds A fragments) and repsT[128][8192]
// (transposed, feeds contiguous async B-tile staging). Zeroes denom and out.
// ---------------------------------------------------------------------------
__global__ __launch_bounds__(256)
void nt_xent_normalize_kernel(const float* __restrict__ teacher,
                              const float* __restrict__ student,
                              _Float16* __restrict__ reps,
                              _Float16* __restrict__ repsT,
                              float* __restrict__ denom,
                              float* __restrict__ out) {
    int tid = blockIdx.x * blockDim.x + threadIdx.x;
    if (tid == 0) *out = 0.0f;
    if (tid < TWO_N) denom[tid] = 0.0f;

    int row  = tid >> 5;
    int lane = tid & 31;
    if (row >= TWO_N) return;

    const float* src = (row < NROW) ? (student + (size_t)row * DIM)
                                    : (teacher + (size_t)(row - NROW) * DIM);
    v4f x = *(const v4f*)(src + lane * 4);
    float ss = x[0]*x[0] + x[1]*x[1] + x[2]*x[2] + x[3]*x[3];
#pragma unroll
    for (int off = 16; off > 0; off >>= 1) ss += __shfl_xor(ss, off, 32);

    float scale = 1.0f / fmaxf(sqrtf(ss), 1e-12f);  // matches F.normalize eps
    v4h o;
    o[0] = (_Float16)(x[0] * scale);
    o[1] = (_Float16)(x[1] * scale);
    o[2] = (_Float16)(x[2] * scale);
    o[3] = (_Float16)(x[3] * scale);
    *(v4h*)(reps + (size_t)row * DIM + lane * 4) = o;
#pragma unroll
    for (int i = 0; i < 4; ++i)
        repsT[(size_t)(lane * 4 + i) * TWO_N + row] = o[i];
}

// ---------------------------------------------------------------------------
// Kernel 2: fused sim = reps @ reps^T -> denom[k] += sum_{l != k} exp(sim[k,l])
// Grid (32 row-blocks, 8 column-splits). 8 waves/block, 32 rows/wave (two
// 16-row tiles, A fragments register-resident for full K=128).
// Per 16-column tile: double-buffered async global->LDS staging of the
// transposed B tile from repsT (contiguous b128 copies, ASYNCcnt), then
// 4 K-chunks x 2 row-tiles of v_wmma_f32_16x16x32_f16, exp fused in-register.
//
// Fragment layouts per CDNA5 ISA 7.12.2 (wave32):
//   A 16x32 f16 : lane half h = lane>>4, row m = lane&15;
//                 elems 0..7  = K[kc*32 + 8h .. +7], elems 8..15 = +16.
//   B 32x16 f16 : column n = lane&15; elems 0..7 = K[kc*32 + 8h .. +7], elems 8..15 = +16
//                 -> 16 scalar ds reads, stride 16 elems, from the [c][n] LDS tile.
//   C/D 16x16 f32: VGPR v of lane -> (M = v + 8h, N = lane&15).
// ---------------------------------------------------------------------------
__global__ __launch_bounds__(256)
void nt_xent_simexp_wmma_kernel(const _Float16* __restrict__ reps,
                                const _Float16* __restrict__ repsT,
                                float* __restrict__ denom) {
    __shared__ _Float16 ldsB[2][16 * DIM];  // two 4 KB transposed B tiles [c][n]

    const int wave    = threadIdx.x >> 5;
    const int lane    = threadIdx.x & 31;
    const int half    = lane >> 4;
    const int l15     = lane & 15;
    const int rowBase = (blockIdx.x * 8 + wave) * 32;
    const int ctBase  = blockIdx.y * TILES_PER_SPLIT;

    // --- A fragments for two 16-row tiles, full K, kept in registers. ---
    v16h a0[4], a1[4];
    const _Float16* ar0 = reps + (size_t)(rowBase + l15) * DIM;
    const _Float16* ar1 = reps + (size_t)(rowBase + 16 + l15) * DIM;
#pragma unroll
    for (int kc = 0; kc < 4; ++kc) {
        v8h lo0 = *(const v8h*)(ar0 + kc * 32 + 8 * half);
        v8h hi0 = *(const v8h*)(ar0 + kc * 32 + 16 + 8 * half);
        v8h lo1 = *(const v8h*)(ar1 + kc * 32 + 8 * half);
        v8h hi1 = *(const v8h*)(ar1 + kc * 32 + 16 + 8 * half);
#pragma unroll
        for (int e = 0; e < 8; ++e) {
            a0[kc][e] = lo0[e]; a0[kc][8 + e] = hi0[e];
            a1[kc][e] = lo1[e]; a1[kc][8 + e] = hi1[e];
        }
    }

    v8f acc0, acc1;
#pragma unroll
    for (int v = 0; v < 8; ++v) { acc0[v] = 0.0f; acc1[v] = 0.0f; }

    // Staging assignment: 256 threads x one async b128 = 4 KB tile.
    const int stg_c   = threadIdx.x >> 1;        // column of reps (0..127)
    const int stg_seg = threadIdx.x & 1;         // 16-byte segment within 32B
    const _Float16* gBase = repsT + (size_t)stg_c * TWO_N + stg_seg * 8;

    // Prologue: stage tile ctBase into buffer 0.
    async_copy_b128(
        (uint32_t)(uintptr_t)&ldsB[0][stg_c * 16 + stg_seg * 8],
        gBase + (size_t)ctBase * 16);
    wait_async_zero();
    __syncthreads();

    for (int t = 0; t < TILES_PER_SPLIT; ++t) {
        const int ct  = ctBase + t;
        const int cur = t & 1;

        // Issue async prefetch of next tile into the other buffer (overlaps
        // with the WMMA work below; completion enforced at iteration end).
        if (t + 1 < TILES_PER_SPLIT) {
            async_copy_b128(
                (uint32_t)(uintptr_t)&ldsB[(t + 1) & 1][stg_c * 16 + stg_seg * 8],
                gBase + (size_t)(ct + 1) * 16);
        }

        v8f c0, c1;
#pragma unroll
        for (int v = 0; v < 8; ++v) { c0[v] = 0.0f; c1[v] = 0.0f; }

#pragma unroll
        for (int kc = 0; kc < 4; ++kc) {
            const _Float16* bp = &ldsB[cur][(kc * 32 + 8 * half) * 16 + l15];
            v16h b;
#pragma unroll
            for (int e = 0; e < 8; ++e) { b[e] = bp[e * 16]; b[8 + e] = bp[(16 + e) * 16]; }
            c0 = __builtin_amdgcn_wmma_f32_16x16x32_f16(
                     false, a0[kc], false, b, (short)0, c0, false, false);
            c1 = __builtin_amdgcn_wmma_f32_16x16x32_f16(
                     false, a1[kc], false, b, (short)0, c1, false, false);
        }

        // exp + diagonal-masked accumulation (tau == 1, |sim| <= 1).
        const int col = ct * 16 + l15;
#pragma unroll
        for (int v = 0; v < 8; ++v) {
            int r0 = rowBase + v + 8 * half;
            int r1 = r0 + 16;
            float e0 = __expf(c0[v]);
            float e1 = __expf(c1[v]);
            acc0[v] += (col == r0) ? 0.0f : e0;
            acc1[v] += (col == r1) ? 0.0f : e1;
        }

        wait_async_zero();   // next tile resident in LDS
        __syncthreads();     // visible to all waves; cur buffer free to overwrite
    }

    // Reduce each row across its 16-lane half (xor 1,2,4,8 stays in-half).
#pragma unroll
    for (int off = 1; off < 16; off <<= 1) {
#pragma unroll
        for (int v = 0; v < 8; ++v) {
            acc0[v] += __shfl_xor(acc0[v], off, 32);
            acc1[v] += __shfl_xor(acc1[v], off, 32);
        }
    }
    if (l15 == 0) {
#pragma unroll
        for (int v = 0; v < 8; ++v) {
            atomicAdd(&denom[rowBase + v + 8 * half], acc0[v]);
            atomicAdd(&denom[rowBase + 16 + v + 8 * half], acc1[v]);
        }
    }
}

// ---------------------------------------------------------------------------
// Kernel 3: positives in fp32 from raw inputs + final mean reduction.
// ---------------------------------------------------------------------------
__global__ __launch_bounds__(256)
void nt_xent_loss_kernel(const float* __restrict__ teacher,
                         const float* __restrict__ student,
                         const float* __restrict__ denom,
                         float* __restrict__ out) {
    int tid  = blockIdx.x * blockDim.x + threadIdx.x;
    int k    = tid >> 5;
    int lane = tid & 31;
    if (k >= NROW) return;

    v4f t = *(const v4f*)(teacher + (size_t)k * DIM + lane * 4);
    v4f s = *(const v4f*)(student + (size_t)k * DIM + lane * 4);
    float dts = 0.0f, nt = 0.0f, ns = 0.0f;
#pragma unroll
    for (int i = 0; i < 4; ++i) {
        dts += t[i] * s[i];
        nt  += t[i] * t[i];
        ns  += s[i] * s[i];
    }
#pragma unroll
    for (int off = 16; off > 0; off >>= 1) {
        dts += __shfl_xor(dts, off, 32);
        nt  += __shfl_xor(nt,  off, 32);
        ns  += __shfl_xor(ns,  off, 32);
    }
    if (lane == 0) {
        float pos = dts / (fmaxf(sqrtf(nt), 1e-12f) * fmaxf(sqrtf(ns), 1e-12f));
        float contrib = (__logf(denom[k]) - pos) + (__logf(denom[k + NROW]) - pos);
        atomicAdd(out, contrib * (1.0f / (2.0f * NROW)));
    }
}

// ---------------------------------------------------------------------------
extern "C" void kernel_launch(void* const* d_in, const int* in_sizes, int n_in,
                              void* d_out, int out_size, void* d_ws, size_t ws_size,
                              hipStream_t stream) {
    const float* teacher = (const float*)d_in[0];
    const float* student = (const float*)d_in[1];
    float* out = (float*)d_out;

    char* ws = (char*)d_ws;
    _Float16* reps  = (_Float16*)ws;                                  // 2 MB
    _Float16* repsT = (_Float16*)(ws + (size_t)TWO_N * DIM * 2);      // 2 MB
    float*    denom = (float*)(ws + (size_t)TWO_N * DIM * 4);         // 32 KB

    nt_xent_normalize_kernel<<<(TWO_N * 32) / 256, 256, 0, stream>>>(
        teacher, student, reps, repsT, denom, out);

    dim3 grid2(ROW_BLOCKS, COL_SPLIT);
    nt_xent_simexp_wmma_kernel<<<grid2, 256, 0, stream>>>(reps, repsT, denom);

    nt_xent_loss_kernel<<<(NROW * 32) / 256, 256, 0, stream>>>(
        teacher, student, denom, out);
}
